// NovaNet_14851996910054
// MI455X (gfx1250) — matrix operand order repaired and checked by reading, not compiled, not run on hardware
//
#include <hip/hip_runtime.h>

#define N_NODES 30000
#define M_EDGES 6000
#define DH 64
#define DIN 256
#define DOUT 128
#define NLAYERS 4
#define NPAD 30016          // N rounded up to 32 (zero-padded f16 hT)
#define MPAD 6016           // M rounded up to 32 (zero-padded f16 eT)
#define NSLICE 8            // split-K slices for edge aggregation
#define SLICE_LEN 3776      // 118*32, NSLICE*SLICE_LEN >= N

typedef __attribute__((ext_vector_type(16))) _Float16 v16h;
typedef __attribute__((ext_vector_type(8)))  _Float16 v8h;
typedef __attribute__((ext_vector_type(8)))  float    v8f;
typedef __attribute__((ext_vector_type(4)))  float    v4f;

static __device__ __forceinline__ v8f zero8() {
  v8f z;
#pragma unroll
  for (int i = 0; i < 8; ++i) z[i] = 0.0f;
  return z;
}

static __device__ __forceinline__ v16h cat16(v8h lo, v8h hi) {
  v16h r;
#pragma unroll
  for (int i = 0; i < 8; ++i) { r[i] = lo[i]; r[8 + i] = hi[i]; }
  return r;
}

static __device__ __forceinline__ v8h cvt8(v4f a, v4f b) {
  v8h r;
#pragma unroll
  for (int i = 0; i < 4; ++i) { r[i] = (_Float16)a[i]; r[4 + i] = (_Float16)b[i]; }
  return r;
}

// A tile (16 rows x 32 K) in f16 WMMA A layout, sourced from a row-major f32 matrix.
// Lane L: row = r0 + L%16; Kbase = (L<16)?0:8; v0..3 <- K=Kbase..Kbase+7, v4..7 <- K=16+Kbase..+7
static __device__ __forceinline__ v16h loadA_f32(const float* base, int lda, int r0, int k0, int lane) {
  const float* p = base + (size_t)(r0 + (lane & 15)) * lda + k0 + ((lane < 16) ? 0 : 8);
  v4f a0 = *(const v4f*)(p);
  v4f a1 = *(const v4f*)(p + 4);
  v4f a2 = *(const v4f*)(p + 16);
  v4f a3 = *(const v4f*)(p + 20);
  return cat16(cvt8(a0, a1), cvt8(a2, a3));
}

// Same, but from a precomputed lane row pointer (already offset by Kbase).
static __device__ __forceinline__ v16h loadA_f32_row(const float* p, int k0) {
  v4f a0 = *(const v4f*)(p + k0);
  v4f a1 = *(const v4f*)(p + k0 + 4);
  v4f a2 = *(const v4f*)(p + k0 + 16);
  v4f a3 = *(const v4f*)(p + k0 + 20);
  return cat16(cvt8(a0, a1), cvt8(a2, a3));
}

// A tile from a row-major f16 matrix (rows = output M dim, cols = K).
static __device__ __forceinline__ v16h loadA_f16(const _Float16* base, int lda, int r0, int k0, int lane) {
  const _Float16* p = base + (size_t)(r0 + (lane & 15)) * lda + k0 + ((lane < 16) ? 0 : 8);
  v8h lo = *(const v8h*)(p);
  v8h hi = *(const v8h*)(p + 16);
  return cat16(lo, hi);
}

// B tile (32 K x 16 cols) where B[k][c] = BT[c][k], BT row-major f16 [ncols][K].
// Lane L: col = c0 + L%16; Kbase = (L<16)?0:8; v0..7 <- K=Kbase..Kbase+7, v8..15 <- K=16+Kbase..+7.
static __device__ __forceinline__ v16h loadB_T16(const _Float16* bt, int ldb, int c0, int k0, int lane) {
  const _Float16* p = bt + (size_t)(c0 + (lane & 15)) * ldb + k0 + ((lane < 16) ? 0 : 8);
  v8h lo = *(const v8h*)(p);
  v8h hi = *(const v8h*)(p + 16);
  return cat16(lo, hi);
}

// B tile for edge aggregation: B[k][m] = H_in[n0+k][m], 16 row-coalesced scalar loads.
static __device__ __forceinline__ v16h loadB_Hin(const float* __restrict__ Hin, int n0, int kb, int m) {
  v16h b;
#pragma unroll
  for (int j = 0; j < 16; ++j)
    b[j] = (_Float16)Hin[(size_t)(n0 + kb + j + (j < 8 ? 0 : 8)) * M_EDGES + m];
  return b;
}

static __device__ __forceinline__ v16h loadB_Hin_guard(const float* __restrict__ Hin, int n0, int kb,
                                                       int m, int nend) {
  v16h b;
#pragma unroll
  for (int j = 0; j < 16; ++j) {
    int r = n0 + kb + j + (j < 8 ? 0 : 8);
    int rc = r < nend ? r : 0;
    float vf = Hin[(size_t)rc * M_EDGES + m];
    b[j] = (_Float16)(r < nend ? vf : 0.0f);
  }
  return b;
}

static __device__ __forceinline__ v8f wmma_f16(v16h a, v16h b, v8f c) {
  return __builtin_amdgcn_wmma_f32_16x16x32_f16(false, a, false, b, (short)0, c, false, false);
}

// ---------------------------------------------------------------------------
// prep: transpose+cvt weights to f16, zero hT16 tail padding
// ---------------------------------------------------------------------------
__global__ void k_prep(const float* __restrict__ W_in, const float* __restrict__ W_out,
                       _Float16* __restrict__ WinT, _Float16* __restrict__ WoutT,
                       _Float16* __restrict__ hT16) {
  int idx = blockIdx.x * blockDim.x + threadIdx.x;
  if (idx < DH * DIN) {                       // WinT[d][k] = W_in[k][d]
    int d = idx / DIN, k = idx % DIN;
    WinT[idx] = (_Float16)W_in[k * DH + d];
  } else if (idx < DH * DIN + DOUT * DH) {    // WoutT[dout][k] = W_out[k][dout]
    int j = idx - DH * DIN;
    int dout = j / DH, k = j % DH;
    WoutT[j] = (_Float16)W_out[k * DOUT + dout];
  } else if (idx < DH * DIN + DOUT * DH + DH * (NPAD - N_NODES)) {
    int j = idx - (DH * DIN + DOUT * DH);
    int d = j / (NPAD - N_NODES), n = N_NODES + j % (NPAD - N_NODES);
    hT16[(size_t)d * NPAD + n] = (_Float16)0.0f;
  }
}

// ---------------------------------------------------------------------------
// input projection: h = relu(x @ W_in + b_in), also write transposed f16 copy
// ---------------------------------------------------------------------------
__global__ void k_in(const float* __restrict__ x, const float* __restrict__ b_in,
                     const _Float16* __restrict__ WinT,
                     float* __restrict__ h, _Float16* __restrict__ hT16) {
  int lane = threadIdx.x & 31;
  int w = blockIdx.x * (blockDim.x >> 5) + (threadIdx.x >> 5);
  if (w >= N_NODES / 16) return;
  int n0 = w * 16;
  v8f acc[4] = {zero8(), zero8(), zero8(), zero8()};
  for (int k0 = 0; k0 < DIN; k0 += 32) {
    v16h a = loadA_f32(x, DIN, n0, k0, lane);
    v16h b0 = loadB_T16(WinT, DIN, 0,  k0, lane);
    v16h b1 = loadB_T16(WinT, DIN, 16, k0, lane);
    v16h b2 = loadB_T16(WinT, DIN, 32, k0, lane);
    v16h b3 = loadB_T16(WinT, DIN, 48, k0, lane);
    acc[0] = wmma_f16(a, b0, acc[0]);
    acc[1] = wmma_f16(a, b1, acc[1]);
    acc[2] = wmma_f16(a, b2, acc[2]);
    acc[3] = wmma_f16(a, b3, acc[3]);
  }
  int rowoff = (lane < 16) ? 0 : 8, cl = lane & 15;
#pragma unroll
  for (int t = 0; t < 4; ++t) {
    int d = t * 16 + cl;
    float bias = b_in[d];
#pragma unroll
    for (int v = 0; v < 8; ++v) {
      int n = n0 + v + rowoff;
      float val = acc[t][v] + bias;
      val = val > 0.0f ? val : 0.0f;
      h[(size_t)n * DH + d] = val;
      hT16[(size_t)d * NPAD + n] = (_Float16)val;
    }
  }
}

// ---------------------------------------------------------------------------
// global context stage 1: partial column sums of h
// ---------------------------------------------------------------------------
__global__ void k_g1(const float* __restrict__ h, float* __restrict__ gpart) {
  __shared__ float sm[256];
  int tid = threadIdx.x;
  int d = tid & 63, rs = tid >> 6;
  float acc = 0.0f;
  for (int r = blockIdx.x * 4 + rs; r < N_NODES; r += 120 * 4)
    acc += h[(size_t)r * DH + d];
  sm[tid] = acc;
  __syncthreads();
  if (tid < 64)
    gpart[blockIdx.x * 64 + tid] = sm[tid] + sm[tid + 64] + sm[tid + 128] + sm[tid + 192];
}

// global context stage 2: g = mean(h) @ W_glob[l] + b_glob[l]
__global__ void k_g2(const float* __restrict__ gpart, const float* __restrict__ W_glob,
                     const float* __restrict__ b_glob, int layer, float* __restrict__ g) {
  __shared__ float mean[64];
  int t = threadIdx.x;  // 64 threads
  float s = 0.0f;
  for (int b = 0; b < 120; ++b) s += gpart[b * 64 + t];
  mean[t] = s * (1.0f / (float)N_NODES);
  __syncthreads();
  float acc = b_glob[layer * DH + t];
  for (int k = 0; k < DH; ++k)
    acc += mean[k] * W_glob[((size_t)layer * DH + k) * DH + t];
  g[t] = acc;
}

// ---------------------------------------------------------------------------
// edge aggregation (split-K, software-pipelined):
// epart[s][d][m] = sum_{n in slice s} h[n][d]*H_in[n][m]
// ---------------------------------------------------------------------------
__global__ void k_emain(const float* __restrict__ Hin, const _Float16* __restrict__ hT16,
                        float* __restrict__ epart) {
  int lane = threadIdx.x & 31;
  int w = blockIdx.x * (blockDim.x >> 5) + (threadIdx.x >> 5);  // 0..2999
  int mt = w % (M_EDGES / 16);
  int s  = w / (M_EDGES / 16);
  int m0 = mt * 16;
  int nstart = s * SLICE_LEN;
  int nend = nstart + SLICE_LEN; if (nend > N_NODES) nend = N_NODES;
  int nfull = (nend - nstart) >> 5;
  int rem = (nend - nstart) & 31;  // 0 or 16
  int m = m0 + (lane & 15);
  int kb = (lane < 16) ? 0 : 8;

  v8f acc[4] = {zero8(), zero8(), zero8(), zero8()};
  int n0 = nstart;

  // chunk 0
  v16h b_cur = loadB_Hin(Hin, n0, kb, m);
  v16h a_cur[4];
#pragma unroll
  for (int t = 0; t < 4; ++t) a_cur[t] = loadA_f16(hT16, NPAD, t * 16, n0, lane);

  // pipelined: compute chunk c while fetching chunk c+1 into fresh registers
  for (int c = 0; c < nfull - 1; ++c) {
    int nn = n0 + 32;
    v16h b_n = loadB_Hin(Hin, nn, kb, m);
    v16h a_n[4];
#pragma unroll
    for (int t = 0; t < 4; ++t) a_n[t] = loadA_f16(hT16, NPAD, t * 16, nn, lane);
#pragma unroll
    for (int t = 0; t < 4; ++t) acc[t] = wmma_f16(a_cur[t], b_cur, acc[t]);
    b_cur = b_n;
#pragma unroll
    for (int t = 0; t < 4; ++t) a_cur[t] = a_n[t];
    n0 = nn;
  }

  if (rem) {  // only last slice; hT16 is zero-padded past N so full A loads stay in-bounds
    int nn = n0 + 32;
    v16h b_t = loadB_Hin_guard(Hin, nn, kb, m, nend);
    v16h a_t[4];
#pragma unroll
    for (int t = 0; t < 4; ++t) a_t[t] = loadA_f16(hT16, NPAD, t * 16, nn, lane);
#pragma unroll
    for (int t = 0; t < 4; ++t) acc[t] = wmma_f16(a_cur[t], b_cur, acc[t]);
#pragma unroll
    for (int t = 0; t < 4; ++t) acc[t] = wmma_f16(a_t[t], b_t, acc[t]);
  } else {
#pragma unroll
    for (int t = 0; t < 4; ++t) acc[t] = wmma_f16(a_cur[t], b_cur, acc[t]);
  }

  int rowoff = (lane < 16) ? 0 : 8, cl = lane & 15;
#pragma unroll
  for (int t = 0; t < 4; ++t)
#pragma unroll
    for (int v = 0; v < 8; ++v)
      epart[((size_t)s * DH + t * 16 + v + rowoff) * MPAD + m0 + cl] = acc[t][v];
}

// ---------------------------------------------------------------------------
// edge epilogue: reduce split-K slices, eT16 = f16(relu(W_edge[l]^T @ e_preT + b_edge[l]))
// ---------------------------------------------------------------------------
__global__ void k_eepi(const float* __restrict__ epart, const float* __restrict__ W_edge,
                       const float* __restrict__ b_edge, int layer, _Float16* __restrict__ eT16) {
  int m = blockIdx.x * blockDim.x + threadIdx.x;
  if (m >= MPAD) return;
  if (m >= M_EDGES) {
    for (int d = 0; d < DH; ++d) eT16[(size_t)d * MPAD + m] = (_Float16)0.0f;
    return;
  }
  float col[DH];
#pragma unroll
  for (int k = 0; k < DH; ++k) {
    float s = 0.0f;
    for (int sl = 0; sl < NSLICE; ++sl)
      s += epart[((size_t)sl * DH + k) * MPAD + m];
    col[k] = s;
  }
  const float* W = W_edge + (size_t)layer * DH * DH;
  const float* bb = b_edge + layer * DH;
  for (int dout = 0; dout < DH; ++dout) {
    float acc = bb[dout];
#pragma unroll
    for (int k = 0; k < DH; ++k) acc += col[k] * W[k * DH + dout];
    acc = acc > 0.0f ? acc : 0.0f;
    eT16[(size_t)dout * MPAD + m] = (_Float16)acc;
  }
}

// ---------------------------------------------------------------------------
// node scatter (software-pipelined + stream prefetch):
// msg_pre[n][d] = sum_m H_out[n][m] * e[m][d]
// ---------------------------------------------------------------------------
__global__ void k_mmain(const float* __restrict__ Hout, const _Float16* __restrict__ eT16,
                        float* __restrict__ msg) {
  int lane = threadIdx.x & 31;
  int w = blockIdx.x * (blockDim.x >> 5) + (threadIdx.x >> 5);
  if (w >= N_NODES / 16) return;
  int n0 = w * 16;
  v8f acc[4] = {zero8(), zero8(), zero8(), zero8()};
  const float* arow = Hout + (size_t)(n0 + (lane & 15)) * M_EDGES + ((lane < 16) ? 0 : 8);

  // chunk 0
  v16h a_cur = loadA_f32_row(arow, 0);
  v16h b_cur[4];
#pragma unroll
  for (int t = 0; t < 4; ++t) b_cur[t] = loadB_T16(eT16, MPAD, t * 16, 0, lane);

  int m0 = 0;
  for (int c = 0; c < M_EDGES / 32 - 1; ++c) {  // compute 0..185, fetch 1..186
    int mn = m0 + 32;
    if (m0 + 256 < M_EDGES) __builtin_prefetch(arow + m0 + 256, 0, 3);
    v16h a_n = loadA_f32_row(arow, mn);
    v16h b_n[4];
#pragma unroll
    for (int t = 0; t < 4; ++t) b_n[t] = loadB_T16(eT16, MPAD, t * 16, mn, lane);
#pragma unroll
    for (int t = 0; t < 4; ++t) acc[t] = wmma_f16(a_cur, b_cur[t], acc[t]);
    a_cur = a_n;
#pragma unroll
    for (int t = 0; t < 4; ++t) b_cur[t] = b_n[t];
    m0 = mn;
  }

  // cur = chunk 186 (m0 = 5952); fetch tail chunk (m0 = 5984, 16 valid K)
  {
    const float* p = arow + 5984;
    v4f a0 = *(const v4f*)(p);
    v4f a1 = *(const v4f*)(p + 4);
    v8h lo = cvt8(a0, a1);
    v8h hz;
#pragma unroll
    for (int i = 0; i < 8; ++i) hz[i] = (_Float16)0.0f;
    v16h a_t = cat16(lo, hz);
    v16h b_t[4];
#pragma unroll
    for (int t = 0; t < 4; ++t) b_t[t] = loadB_T16(eT16, MPAD, t * 16, 5984, lane);
#pragma unroll
    for (int t = 0; t < 4; ++t) acc[t] = wmma_f16(a_cur, b_cur[t], acc[t]);
#pragma unroll
    for (int t = 0; t < 4; ++t) acc[t] = wmma_f16(a_t, b_t[t], acc[t]);
  }

  int rowoff = (lane < 16) ? 0 : 8, cl = lane & 15;
#pragma unroll
  for (int t = 0; t < 4; ++t)
#pragma unroll
    for (int v = 0; v < 8; ++v)
      msg[(size_t)(n0 + v + rowoff) * DH + t * 16 + cl] = acc[t][v];
}

// ---------------------------------------------------------------------------
// node epilogue: h = relu(msg_pre @ W_node[l] + b_node[l] + g) + h ; refresh hT16
// ---------------------------------------------------------------------------
__global__ void k_mepi(const float* __restrict__ msg, const float* __restrict__ W_node,
                       const float* __restrict__ b_node, const float* __restrict__ g,
                       int layer, float* __restrict__ h, _Float16* __restrict__ hT16) {
  __shared__ float rb[4][DH];
  int tid = threadIdx.x;
  int nb = blockIdx.x * 4;
  int r = tid >> 6, dout = tid & 63;
  int nload = nb + r;
  rb[r][dout] = (nload < N_NODES) ? msg[(size_t)nload * DH + dout] : 0.0f;
  __syncthreads();
  int n = nb + r;
  if (n >= N_NODES) return;
  const float* W = W_node + (size_t)layer * DH * DH;
  float acc = b_node[layer * DH + dout] + g[dout];
#pragma unroll 8
  for (int k = 0; k < DH; ++k) acc += rb[r][k] * W[k * DH + dout];
  float hn = (acc > 0.0f ? acc : 0.0f) + h[(size_t)n * DH + dout];
  h[(size_t)n * DH + dout] = hn;
  hT16[(size_t)dout * NPAD + n] = (_Float16)hn;
}

// ---------------------------------------------------------------------------
// output head: logits = h @ W_out + b_out  (WMMA, 8 col-tiles)
// ---------------------------------------------------------------------------
__global__ void k_out(const float* __restrict__ h, const _Float16* __restrict__ WoutT,
                      const float* __restrict__ b_out, float* __restrict__ out) {
  int lane = threadIdx.x & 31;
  int w = blockIdx.x * (blockDim.x >> 5) + (threadIdx.x >> 5);
  if (w >= N_NODES / 16) return;
  int n0 = w * 16;
  v8f acc[8] = {zero8(), zero8(), zero8(), zero8(), zero8(), zero8(), zero8(), zero8()};
  for (int k0 = 0; k0 < DH; k0 += 32) {
    v16h a = loadA_f32(h, DH, n0, k0, lane);
#pragma unroll
    for (int t = 0; t < 8; ++t) {
      v16h b = loadB_T16(WoutT, DH, t * 16, k0, lane);
      acc[t] = wmma_f16(a, b, acc[t]);
    }
  }
  int rowoff = (lane < 16) ? 0 : 8, cl = lane & 15;
#pragma unroll
  for (int t = 0; t < 8; ++t) {
    int d = t * 16 + cl;
    float bias = b_out[d];
#pragma unroll
    for (int v = 0; v < 8; ++v)
      out[(size_t)(n0 + v + rowoff) * DOUT + d] = acc[t][v] + bias;
  }
}

// embeddings copy into second output segment
__global__ void k_emb(const float* __restrict__ h, float* __restrict__ out) {
  size_t idx = (size_t)blockIdx.x * blockDim.x + threadIdx.x;
  if (idx < (size_t)N_NODES * DH)
    out[(size_t)N_NODES * DOUT + idx] = h[idx];
}

// ---------------------------------------------------------------------------
extern "C" void kernel_launch(void* const* d_in, const int* in_sizes, int n_in,
                              void* d_out, int out_size, void* d_ws, size_t ws_size,
                              hipStream_t stream) {
  (void)in_sizes; (void)n_in; (void)out_size; (void)ws_size;
  const float* x      = (const float*)d_in[0];
  const float* Hin    = (const float*)d_in[1];
  const float* Hout   = (const float*)d_in[2];
  const float* W_in   = (const float*)d_in[3];
  const float* b_in   = (const float*)d_in[4];
  const float* W_edge = (const float*)d_in[5];
  const float* b_edge = (const float*)d_in[6];
  const float* W_node = (const float*)d_in[7];
  const float* b_node = (const float*)d_in[8];
  const float* W_glob = (const float*)d_in[9];
  const float* b_glob = (const float*)d_in[10];
  const float* W_out  = (const float*)d_in[11];
  const float* b_out  = (const float*)d_in[12];

  char* ws = (char*)d_ws;                                  // all offsets 256B aligned
  float*    h     = (float*)(ws + 0);                      // 7,680,000 B
  _Float16* hT16  = (_Float16*)(ws + 7680000);             // 3,842,048 B
  float*    msg   = (float*)(ws + 11522048);               // 7,680,000 B
  float*    epart = (float*)(ws + 19202048);               // 12,320,768 B
  _Float16* eT16  = (_Float16*)(ws + 31522816);            // 770,048 B
  float*    gpart = (float*)(ws + 32292864);               // 30,720 B
  float*    g     = (float*)(ws + 32323584);               // 256 B
  _Float16* WinT  = (_Float16*)(ws + 32323840);            // 32,768 B
  _Float16* WoutT = (_Float16*)(ws + 32356608);            // 16,384 B
  float* out = (float*)d_out;

  k_prep<<<100, 256, 0, stream>>>(W_in, W_out, WinT, WoutT, hT16);
  k_in<<<(N_NODES / 16 + 3) / 4, 128, 0, stream>>>(x, b_in, WinT, h, hT16);

  for (int l = 0; l < NLAYERS; ++l) {
    k_g1<<<120, 256, 0, stream>>>(h, gpart);
    k_g2<<<1, 64, 0, stream>>>(gpart, W_glob, b_glob, l, g);
    k_emain<<<(M_EDGES / 16) * NSLICE / 4, 128, 0, stream>>>(Hin, hT16, epart);
    k_eepi<<<(MPAD + 255) / 256, 256, 0, stream>>>(epart, W_edge, b_edge, l, eT16);
    k_mmain<<<(N_NODES / 16 + 3) / 4, 128, 0, stream>>>(Hout, eT16, msg);
    k_mepi<<<N_NODES / 4, 256, 0, stream>>>(msg, W_node, b_node, g, l, h, hT16);
  }

  k_out<<<(N_NODES / 16 + 3) / 4, 128, 0, stream>>>(h, WoutT, b_out, out);
  k_emb<<<N_NODES * DH / 256, 256, 0, stream>>>(h, out);
}
